// DiscV_11141145165889
// MI455X (gfx1250) — matrix operand order repaired and checked by reading, not compiled, not run on hardware
//
#include <hip/hip_runtime.h>
#include <hip/hip_bf16.h>
#include <math.h>

// ---------------------------------------------------------------------------
// CDNA5 (gfx1250, wave32). GEMMs on v_wmma_f32_16x16x32_bf16.
// Fragment layouts (contiguous k order used here; D per ISA 7.12.2), lane L, g=L>>4:
//   A (16x32): slots 0..7  = K kb+g*16..+7  (run of 8 consecutive)
//              slots 8..15 = K kb+g*16+8..+15 (run of 8 consecutive)
//   B (32x16): slots 0..15 = K kb+g*16..+15 (16 consecutive, N = lane&15)
//   D (16x16): VGPR r -> M = r+g*8, N = lane&15
// B is pre-converted to bf16 [n][k] -> fragment = one aligned 32B v16bf load.
// Each wave computes a 16x64 strip: 1 A fragment -> 4 WMMAs.
// k_embed_gemm stages its (block-shared) A tile through LDS with
// double-buffered GLOBAL_LOAD_ASYNC_TO_LDS_B128 (ASYNCcnt-tracked).
// ---------------------------------------------------------------------------

typedef __bf16 v16bf __attribute__((ext_vector_type(16)));
typedef float  v8f   __attribute__((ext_vector_type(8)));
typedef int    v4i   __attribute__((ext_vector_type(4)));

#define DIMC 512
#define SEQ  32
#define BS   128
#define VOCAB 10000
#define KCONV (DIMC * 5)
#define PROP 10
#define NTOP 6

#define WMMA_BF16(a, b, c) \
    __builtin_amdgcn_wmma_f32_16x16x32_bf16(false, (a), false, (b), (short)0, (c), false, false)

#if defined(__has_builtin)
#if __has_builtin(__builtin_amdgcn_global_load_async_to_lds_b128)
#define HAVE_ASYNC_LDS 1
#endif
#endif

#ifndef HAVE_ASYNC_LDS
#define HAVE_ASYNC_LDS 0
#endif

#if HAVE_ASYNC_LDS
typedef __attribute__((address_space(1))) v4i gv4i;   // global (device) v4i
typedef __attribute__((address_space(3))) v4i lv4i;   // LDS v4i
__device__ __forceinline__ void async_copy_b128(const float* g, float* l)
{
    __builtin_amdgcn_global_load_async_to_lds_b128((gv4i*)(v4i*)(float*)g,
                                                   (lv4i*)(v4i*)l, 0, 0);
}
__device__ __forceinline__ void wait_async_le1()
{
#if __has_builtin(__builtin_amdgcn_s_wait_asynccnt)
    __builtin_amdgcn_s_wait_asynccnt(1);
#else
    asm volatile("s_wait_asynccnt 0x1" ::: "memory");
#endif
}
__device__ __forceinline__ void wait_async_le0()
{
#if __has_builtin(__builtin_amdgcn_s_wait_asynccnt)
    __builtin_amdgcn_s_wait_asynccnt(0);
#else
    asm volatile("s_wait_asynccnt 0x0" ::: "memory");
#endif
}
#endif

// ------------------------- weight pre-conversion ---------------------------
__global__ void k_cvt_bf16(const float* __restrict__ in, __bf16* __restrict__ out, int n)
{
    for (int i = blockIdx.x * 256 + threadIdx.x; i < n; i += gridDim.x * 256)
        out[i] = (__bf16)in[i];
}

// res_w (o, i, t) -> bf16 (o, k) with k = t*512 + i
__global__ void k_cvt_convw(const float* __restrict__ in, __bf16* __restrict__ out)
{
    const int idx = blockIdx.x * 256 + threadIdx.x;     // 512*2560 total
    if (idx < DIMC * KCONV) {
        const int o = idx / KCONV, k = idx % KCONV;
        const int t = k >> 9, i = k & (DIMC - 1);
        out[idx] = (__bf16)in[((size_t)o * DIMC + i) * 5 + t];
    }
}

// ---------------------------------------------------------------------------
// Kernel 1: x = inputs @ conv_w^T + conv_b   (M=4096, N=512, K=10000)
// A tile (16 rows x 32 k fp32 = 2KB) is shared by all 4 waves -> LDS-staged
// via double-buffered async copy; B strips are per-wave direct v16bf loads.
// ---------------------------------------------------------------------------
__global__ void k_embed_gemm(const float* __restrict__ A,    // (4096,10000) f32
                             const __bf16* __restrict__ Wb,  // (512,10000) bf16
                             const float* __restrict__ bias,
                             float* __restrict__ C)
{
    const int K = VOCAB, N = DIMC;
    const int tid  = threadIdx.x;
    const int lane = tid & 31;
    const int wave = tid >> 5;
    const int m0 = blockIdx.x * 16;
    const int n0 = (blockIdx.y * 4 + wave) * 64;     // 16x64 strip per wave
    const int g = lane >> 4, mr = lane & 15, nc = lane & 15;

    const float*  Arow = A + (size_t)(m0 + mr) * K;
    const __bf16* W0 = Wb + (size_t)(n0 + 0 * 16 + nc) * K;
    const __bf16* W1 = Wb + (size_t)(n0 + 1 * 16 + nc) * K;
    const __bf16* W2 = Wb + (size_t)(n0 + 2 * 16 + nc) * K;
    const __bf16* W3 = Wb + (size_t)(n0 + 3 * 16 + nc) * K;

    v8f acc0 = {}, acc1 = {}, acc2 = {}, acc3 = {};
    const int KMAIN = 9984;                           // 312 * 32
    const int NIT   = KMAIN / 32;

#if HAVE_ASYNC_LDS
    __shared__ __align__(32) float Abuf[2][16][32];   // double-buffered A slab
    // producer mapping: each of the 128 threads copies one 16B b128 chunk
    const int crow = tid >> 3;                        // 0..15
    const int cseg = (tid & 7) * 4;                   // 0,4,...,28
    const float* gsrc = A + (size_t)(m0 + crow) * K + cseg;

    async_copy_b128(gsrc + 0 * 32, &Abuf[0][crow][cseg]);   // it 0
    async_copy_b128(gsrc + 1 * 32, &Abuf[1][crow][cseg]);   // it 1

    for (int it = 0; it < NIT; ++it) {
        const int kb  = it * 32;
        const int buf = it & 1;
        wait_async_le1();                 // oldest outstanding copy complete
        __syncthreads();                  // all waves' copies visible
        const v8f r0 = *(const v8f*)&Abuf[buf][mr][g * 16];
        const v8f r1 = *(const v8f*)&Abuf[buf][mr][g * 16 + 8];
        v16bf a;
#pragma unroll
        for (int e = 0; e < 8; ++e) { a[e] = (__bf16)r0[e]; a[8 + e] = (__bf16)r1[e]; }
        const v16bf b0 = *(const v16bf*)(W0 + kb + g * 16);
        const v16bf b1 = *(const v16bf*)(W1 + kb + g * 16);
        const v16bf b2 = *(const v16bf*)(W2 + kb + g * 16);
        const v16bf b3 = *(const v16bf*)(W3 + kb + g * 16);
        acc0 = WMMA_BF16(a, b0, acc0);
        acc1 = WMMA_BF16(a, b1, acc1);
        acc2 = WMMA_BF16(a, b2, acc2);
        acc3 = WMMA_BF16(a, b3, acc3);
        __syncthreads();                  // done reading buf before reuse
        if (it + 2 < NIT)
            async_copy_b128(gsrc + (size_t)(it + 2) * 32, &Abuf[buf][crow][cseg]);
    }
    wait_async_le0();
#else
    for (int kb = 0; kb < KMAIN; kb += 32) {
        __builtin_prefetch(Arow + kb + 256, 0, 0);
        const v8f r0 = *(const v8f*)(Arow + kb + g * 16);
        const v8f r1 = *(const v8f*)(Arow + kb + g * 16 + 8);
        v16bf a;
#pragma unroll
        for (int e = 0; e < 8; ++e) { a[e] = (__bf16)r0[e]; a[8 + e] = (__bf16)r1[e]; }
        const v16bf b0 = *(const v16bf*)(W0 + kb + g * 16);
        const v16bf b1 = *(const v16bf*)(W1 + kb + g * 16);
        const v16bf b2 = *(const v16bf*)(W2 + kb + g * 16);
        const v16bf b3 = *(const v16bf*)(W3 + kb + g * 16);
        acc0 = WMMA_BF16(a, b0, acc0);
        acc1 = WMMA_BF16(a, b1, acc1);
        acc2 = WMMA_BF16(a, b2, acc2);
        acc3 = WMMA_BF16(a, b3, acc3);
    }
#endif
    {   // K tail: k in [9984,10000). A lanes g=1 hold k>=10000 -> select 0.
        // B: lanes g=1 would read k>=10000 -> load valid g=0 strip, select 0.
        const int kb = KMAIN;
        const v8f r0 = *(const v8f*)(Arow + kb);        // always valid
        const v8f r1 = *(const v8f*)(Arow + kb + 8);
        v16bf a = {};
#pragma unroll
        for (int e = 0; e < 8; ++e) { a[e] = g ? (__bf16)0.f : (__bf16)r0[e]; a[8 + e] = g ? (__bf16)0.f : (__bf16)r1[e]; }
        const v16bf bz = {};
        const v16bf l0 = *(const v16bf*)(W0 + kb);
        const v16bf l1 = *(const v16bf*)(W1 + kb);
        const v16bf l2 = *(const v16bf*)(W2 + kb);
        const v16bf l3 = *(const v16bf*)(W3 + kb);
        const v16bf b0 = g ? bz : l0;
        const v16bf b1 = g ? bz : l1;
        const v16bf b2 = g ? bz : l2;
        const v16bf b3 = g ? bz : l3;
        acc0 = WMMA_BF16(a, b0, acc0);
        acc1 = WMMA_BF16(a, b1, acc1);
        acc2 = WMMA_BF16(a, b2, acc2);
        acc3 = WMMA_BF16(a, b3, acc3);
    }
#pragma unroll
    for (int r = 0; r < 8; ++r) {
        const int m = m0 + r + g * 8;
        C[(size_t)m * N + n0 +  0 + nc] = acc0[r] + bias[n0 +  0 + nc];
        C[(size_t)m * N + n0 + 16 + nc] = acc1[r] + bias[n0 + 16 + nc];
        C[(size_t)m * N + n0 + 32 + nc] = acc2[r] + bias[n0 + 32 + nc];
        C[(size_t)m * N + n0 + 48 + nc] = acc3[r] + bias[n0 + 48 + nc];
    }
}

// ---------------------------------------------------------------------------
// Kernel 2: conv1d(relu(X), W, b) k=5 pad=2, implicit GEMM (K=2560).
//   A(m=b*32+s, k=t*512+i) = relu(X[b][s+t-2][i]) with zero seq-padding,
//   realized branch-free: clamp row, select-to-zero on validity.
//   res: nullptr -> Y = conv ; else Y = res + 0.3*conv (in-place safe).
// ---------------------------------------------------------------------------
__global__ void k_conv_gemm(const float* __restrict__ X,
                            const __bf16* __restrict__ Wt,  // (512, 2560) bf16
                            const float* __restrict__ bias,
                            const float* __restrict__ res,
                            float* __restrict__ Y)
{
    const int N = DIMC;
    const int lane = threadIdx.x & 31;
    const int wave = threadIdx.x >> 5;
    const int m0 = blockIdx.x * 16;
    const int n0 = (blockIdx.y * 4 + wave) * 64;
    const int g = lane >> 4, mr = lane & 15, nc = lane & 15;

    const int m = m0 + mr;          // m = b*SEQ + s
    const int bb = m >> 5;
    const int ss = m & 31;
    const float* Xb = X + (size_t)bb * SEQ * DIMC;
    const __bf16* W0 = Wt + (size_t)(n0 + 0 * 16 + nc) * KCONV;
    const __bf16* W1 = Wt + (size_t)(n0 + 1 * 16 + nc) * KCONV;
    const __bf16* W2 = Wt + (size_t)(n0 + 2 * 16 + nc) * KCONV;
    const __bf16* W3 = Wt + (size_t)(n0 + 3 * 16 + nc) * KCONV;

    v8f acc0 = {}, acc1 = {}, acc2 = {}, acc3 = {};
    for (int kb = 0; kb < KCONV; kb += 32) {
        v16bf a;
#pragma unroll
        for (int run = 0; run < 2; ++run) {           // two runs of 8 consecutive k
            const int k0 = kb + g * 16 + run * 8;     // t constant within run
            const int t  = k0 >> 9;
            const int i0 = k0 & (DIMC - 1);
            const int sp = ss + t - 2;
            const int spc = sp < 0 ? 0 : (sp > SEQ - 1 ? SEQ - 1 : sp);
            const bool valid = (sp >= 0) && (sp < SEQ);
            const v8f r = *(const v8f*)(Xb + (size_t)spc * DIMC + i0);
#pragma unroll
            for (int e = 0; e < 8; ++e) {
                const float f = valid ? fmaxf(r[e], 0.f) : 0.f;   // relu + pad
                a[run * 8 + e] = (__bf16)f;
            }
        }
        const v16bf b0 = *(const v16bf*)(W0 + kb + g * 16);
        const v16bf b1 = *(const v16bf*)(W1 + kb + g * 16);
        const v16bf b2 = *(const v16bf*)(W2 + kb + g * 16);
        const v16bf b3 = *(const v16bf*)(W3 + kb + g * 16);
        acc0 = WMMA_BF16(a, b0, acc0);
        acc1 = WMMA_BF16(a, b1, acc1);
        acc2 = WMMA_BF16(a, b2, acc2);
        acc3 = WMMA_BF16(a, b3, acc3);
    }
#pragma unroll
    for (int r = 0; r < 8; ++r) {
        const int mo = m0 + r + g * 8;
#pragma unroll
        for (int nt = 0; nt < 4; ++nt) {
            const int n = n0 + nt * 16 + nc;
            const float v = (nt == 0 ? acc0[r] : nt == 1 ? acc1[r] : nt == 2 ? acc2[r] : acc3[r])
                            + bias[n];
            if (res) Y[(size_t)mo * N + n] = res[(size_t)mo * N + n] + 0.3f * v;
            else     Y[(size_t)mo * N + n] = v;
        }
    }
}

// ---------------------------------------------------------------------------
// Kernel 3: psl_all = concat(obj,motion) @ psl_w^T + psl_b (M=2560,K=1024)
// ---------------------------------------------------------------------------
__global__ void k_psl_gemm(const float* __restrict__ OBJ,
                           const float* __restrict__ MOT,
                           const __bf16* __restrict__ Wb,   // (512,1024) bf16
                           const float* __restrict__ bias,
                           float* __restrict__ C)
{
    const int K = 1024, N = DIMC;
    const int lane = threadIdx.x & 31;
    const int wave = threadIdx.x >> 5;
    const int m0 = blockIdx.x * 16;
    const int n0 = (blockIdx.y * 4 + wave) * 64;
    const int g = lane >> 4, mr = lane & 15, nc = lane & 15;

    const int m = m0 + mr;
    const int bb = m / (2 * PROP);
    const int p  = m % (2 * PROP);
    const float* Arow = (p < PROP)
        ? OBJ + ((size_t)bb * PROP + p) * K
        : MOT + ((size_t)bb * PROP + (p - PROP)) * K;
    const __bf16* W0 = Wb + (size_t)(n0 + 0 * 16 + nc) * K;
    const __bf16* W1 = Wb + (size_t)(n0 + 1 * 16 + nc) * K;
    const __bf16* W2 = Wb + (size_t)(n0 + 2 * 16 + nc) * K;
    const __bf16* W3 = Wb + (size_t)(n0 + 3 * 16 + nc) * K;

    v8f acc0 = {}, acc1 = {}, acc2 = {}, acc3 = {};
    for (int kb = 0; kb < K; kb += 32) {
        const v8f r0 = *(const v8f*)(Arow + kb + g * 16);
        const v8f r1 = *(const v8f*)(Arow + kb + g * 16 + 8);
        v16bf a;
#pragma unroll
        for (int e = 0; e < 8; ++e) { a[e] = (__bf16)r0[e]; a[8 + e] = (__bf16)r1[e]; }
        const v16bf b0 = *(const v16bf*)(W0 + kb + g * 16);
        const v16bf b1 = *(const v16bf*)(W1 + kb + g * 16);
        const v16bf b2 = *(const v16bf*)(W2 + kb + g * 16);
        const v16bf b3 = *(const v16bf*)(W3 + kb + g * 16);
        acc0 = WMMA_BF16(a, b0, acc0);
        acc1 = WMMA_BF16(a, b1, acc1);
        acc2 = WMMA_BF16(a, b2, acc2);
        acc3 = WMMA_BF16(a, b3, acc3);
    }
#pragma unroll
    for (int r = 0; r < 8; ++r) {
        const int mo = m0 + r + g * 8;
        C[(size_t)mo * N + n0 +  0 + nc] = acc0[r] + bias[n0 +  0 + nc];
        C[(size_t)mo * N + n0 + 16 + nc] = acc1[r] + bias[n0 + 16 + nc];
        C[(size_t)mo * N + n0 + 32 + nc] = acc2[r] + bias[n0 + 32 + nc];
        C[(size_t)mo * N + n0 + 48 + nc] = acc3[r] + bias[n0 + 48 + nc];
    }
}

// ---------------------------------------------------------------------------
// Kernel 4: top-6 of sum_s(alpha[b][s][j]*mask[b][s]); '>' = lowest-idx ties.
// ---------------------------------------------------------------------------
__global__ void k_topk(const float* __restrict__ alpha,
                       const float* __restrict__ mask,
                       int* __restrict__ topk)
{
    __shared__ float sums[32];
    const int b = blockIdx.x, t = threadIdx.x;
    float s = -INFINITY;
    if (t < 2 * PROP) {
        s = 0.f;
        for (int q = 0; q < SEQ; ++q)
            s += alpha[((size_t)b * SEQ + q) * (2 * PROP) + t] * mask[b * SEQ + q];
    }
    sums[t] = s;
    __syncthreads();
    if (t == 0) {
        for (int k = 0; k < NTOP; ++k) {
            int best = 0; float bv = -INFINITY;
            for (int j = 0; j < 2 * PROP; ++j)
                if (sums[j] > bv) { bv = sums[j]; best = j; }
            topk[b * NTOP + k] = best;
            sums[best] = -INFINITY;
        }
    }
}

// ---------------------------------------------------------------------------
// Kernel 5: fused tail, one block (8 waves) per batch.
// ---------------------------------------------------------------------------
__global__ void k_tail(const float* __restrict__ att,   // (bs,32,512)
                       const float* __restrict__ psl,   // (bs,20,512)
                       const int* __restrict__ topk,    // (bs,6)
                       const float* __restrict__ mask,  // (bs,1,32)
                       const float* __restrict__ ln_g, const float* __restrict__ ln_b,
                       const float* __restrict__ sv_w, const float* __restrict__ sv_b,
                       const float* __restrict__ ss_w, const float* __restrict__ ss_b,
                       const float* __restrict__ cl_w, const float* __restrict__ cl_b,
                       float* __restrict__ out)
{
    const int b = blockIdx.x, tid = threadIdx.x;
    const int wave = tid >> 5, lane = tid & 31;

    __shared__ float pt [NTOP][DIMC];
    __shared__ float agg[NTOP][DIMC];
    __shared__ float vv [NTOP][DIMC];
    __shared__ float sv [NTOP][DIMC];
    __shared__ float adj[SEQ][NTOP];

    for (int idx = tid; idx < NTOP * DIMC; idx += 256) {
        const int k = idx >> 9, d = idx & (DIMC - 1);
        const int p = topk[b * NTOP + k];
        pt[k][d] = psl[((size_t)b * 2 * PROP + p) * DIMC + d];
    }
    __syncthreads();

    if (tid < SEQ * NTOP) {
        const int s = tid / NTOP, k = tid % NTOP;
        float acc = 0.f;
        const float* ar = att + ((size_t)b * SEQ + s) * DIMC;
        for (int d = 0; d < DIMC; ++d) acc += ar[d] * pt[k][d];
        acc *= 0.044194173824159216f;                 // 1/sqrt(512)
        adj[s][k] = (mask[b * SEQ + s] > 0.f) ? acc : -9.0e15f;
    }
    __syncthreads();

    if (tid < NTOP) {                                 // softmax over seq
        const int k = tid;
        float mx = -INFINITY;
        for (int s = 0; s < SEQ; ++s) mx = fmaxf(mx, adj[s][k]);
        float sum = 0.f;
        for (int s = 0; s < SEQ; ++s) { float e = __expf(adj[s][k] - mx); adj[s][k] = e; sum += e; }
        const float inv = 1.f / sum;
        for (int s = 0; s < SEQ; ++s) adj[s][k] *= inv;
    }
    __syncthreads();

    for (int idx = tid; idx < NTOP * DIMC; idx += 256) {  // psl_agg
        const int k = idx >> 9, d = idx & (DIMC - 1);
        float acc = 0.f;
        for (int s = 0; s < SEQ; ++s)
            acc += att[((size_t)b * SEQ + s) * DIMC + d] * adj[s][k];
        agg[k][d] = acc;
    }
    __syncthreads();

    if (wave < NTOP) {                                // layernorm, wave/row
        float sum = 0.f, sq = 0.f;
        for (int d = lane; d < DIMC; d += 32) { const float v = agg[wave][d]; sum += v; sq += v * v; }
        for (int off = 16; off > 0; off >>= 1) {
            sum += __shfl_xor(sum, off, 32);
            sq  += __shfl_xor(sq,  off, 32);
        }
        const float mu  = sum * (1.f / DIMC);
        const float var = sq * (1.f / DIMC) - mu * mu;
        const float inv = rsqrtf(var + 1e-5f);
        for (int d = lane; d < DIMC; d += 32)
            agg[wave][d] = (agg[wave][d] - mu) * inv * ln_g[d] + ln_b[d];
    }
    __syncthreads();

    for (int idx = tid; idx < NTOP * DIMC; idx += 256) {  // v, s projections
        const int k = idx >> 9, d = idx & (DIMC - 1);
        float av = sv_b[d], as = ss_b[d];
        const float* wv = sv_w + (size_t)d * DIMC;
        const float* ws = ss_w + (size_t)d * DIMC;
        for (int c = 0; c < DIMC; ++c) { av += pt[k][c] * wv[c]; as += agg[k][c] * ws[c]; }
        vv[k][d] = tanhf(av);
        sv[k][d] = tanhf(as);
    }
    __syncthreads();

    if (wave < NTOP) {                                // classifier, wave/k
        float acc = 0.f;
        for (int d = lane; d < DIMC; d += 32) acc += vv[wave][d] * sv[wave][d] * cl_w[d];
        for (int off = 16; off > 0; off >>= 1) acc += __shfl_xor(acc, off, 32);
        if (lane == 0)
            out[b * NTOP + wave] = 1.f / (1.f + __expf(-(acc + cl_b[0])));
    }
}

// ---------------------------------------------------------------------------
extern "C" void kernel_launch(void* const* d_in, const int* in_sizes, int n_in,
                              void* d_out, int out_size, void* d_ws, size_t ws_size,
                              hipStream_t stream)
{
    const float* inputs  = (const float*)d_in[0];
    const float* obj     = (const float*)d_in[2];
    const float* motion  = (const float*)d_in[3];
    const float* attmask = (const float*)d_in[4];
    const float* alpha   = (const float*)d_in[5];
    const float* conv_w  = (const float*)d_in[6];
    const float* conv_b  = (const float*)d_in[7];
    const float* res_w1  = (const float*)d_in[8];
    const float* res_b1  = (const float*)d_in[9];
    const float* res_w2  = (const float*)d_in[10];
    const float* res_b2  = (const float*)d_in[11];
    const float* psl_w   = (const float*)d_in[12];
    const float* psl_b   = (const float*)d_in[13];
    const float* ln_g    = (const float*)d_in[14];
    const float* ln_b    = (const float*)d_in[15];
    const float* sv_w    = (const float*)d_in[16];
    const float* sv_b    = (const float*)d_in[17];
    const float* ss_w    = (const float*)d_in[18];
    const float* ss_b    = (const float*)d_in[19];
    const float* cl_w    = (const float*)d_in[20];
    const float* cl_b    = (const float*)d_in[21];
    float* out = (float*)d_out;

    // -------- workspace layout (all offsets 256B aligned) --------
    char* ws = (char*)d_ws;
    size_t off = 0;
    auto take = [&](size_t bytes) { char* p = ws + off; off += (bytes + 255) & ~(size_t)255; return p; };
    float*  x0   = (float*) take((size_t)BS * SEQ * DIMC * 4);        // x / att_out (8MB)
    float*  y1   = (float*) take((size_t)BS * SEQ * DIMC * 4);        // conv1 out   (8MB)
    float*  psl  = (float*) take((size_t)BS * 2 * PROP * DIMC * 4);   // psl_all     (5MB)
    int*    topk = (int*)   take((size_t)BS * NTOP * 4);
    __bf16* wEb  = (__bf16*)take((size_t)DIMC * VOCAB * 2);           // conv_w bf16 (10MB)
    __bf16* w1t  = (__bf16*)take((size_t)DIMC * KCONV * 2);           // res_w1 bf16 (2.5MB)
    __bf16* w2t  = (__bf16*)take((size_t)DIMC * KCONV * 2);           // res_w2 bf16 (2.5MB)
    __bf16* wPb  = (__bf16*)take((size_t)DIMC * 1024 * 2);            // psl_w bf16  (1MB)

    // -------- weight conversion (bf16, [n][k] layouts) --------
    k_cvt_bf16 <<<2048, 256, 0, stream>>>(conv_w, wEb, DIMC * VOCAB);
    k_cvt_convw<<<(DIMC * KCONV + 255) / 256, 256, 0, stream>>>(res_w1, w1t);
    k_cvt_convw<<<(DIMC * KCONV + 255) / 256, 256, 0, stream>>>(res_w2, w2t);
    k_cvt_bf16 <<<1024, 256, 0, stream>>>(psl_w, wPb, DIMC * 1024);

    const dim3 blk(128);   // 4 waves; each wave owns a 16x64 WMMA strip
    // x0 = inputs @ conv_w^T + conv_b
    k_embed_gemm<<<dim3((BS * SEQ) / 16, 2), blk, 0, stream>>>(inputs, wEb, conv_b, x0);
    // y1 = conv1d(relu(x0), w1) + b1
    k_conv_gemm<<<dim3((BS * SEQ) / 16, 2), blk, 0, stream>>>(x0, w1t, res_b1, nullptr, y1);
    // x0 = x0 + 0.3*(conv1d(relu(y1), w2) + b2)   (in-place residual)
    k_conv_gemm<<<dim3((BS * SEQ) / 16, 2), blk, 0, stream>>>(y1, w2t, res_b2, x0, x0);
    // psl_all = concat(obj,motion) @ psl_w^T + psl_b
    k_psl_gemm<<<dim3((BS * 2 * PROP) / 16, 2), blk, 0, stream>>>(obj, motion, wPb, psl_b, psl);
    // top-6 proposals per batch
    k_topk<<<BS, 32, 0, stream>>>(alpha, attmask, topk);
    // fused attention/softmax/layernorm/classifier tail
    k_tail<<<BS, 256, 0, stream>>>(x0, psl, topk, attmask, ln_g, ln_b,
                                   sv_w, sv_b, ss_w, ss_b, cl_w, cl_b, out);
}
